// _TransformerBlock_90787018703067
// MI455X (gfx1250) — matrix operand order repaired and checked by reading, not compiled, not run on hardware
//
#include <hip/hip_runtime.h>
#include <hip/hip_bf16.h>

// ---------------------------------------------------------------------------
// Transformer block on gfx1250 (MI455X): all matmuls on v_wmma_f32_16x16x32_bf16
// (bf16 in, fp32 accumulate); LayerNorm / softmax / GELU / residuals in fp32.
// GEMMs use 128x128 block tiles, pre-transposed bf16 weights, and
// double-buffered LDS fed by GLOBAL_LOAD_ASYNC_TO_LDS_B128 (ASYNCcnt).
// ---------------------------------------------------------------------------

typedef __attribute__((ext_vector_type(16))) __bf16 v16bf;
typedef __attribute__((ext_vector_type(8)))  float  v8f;

union BF16x16 { v16bf v; unsigned u[8]; };

#define DIM    768
#define NH     12
#define HD     64
#define NCTX   2048
#define HIDDEN 3072

#define BM 128
#define BN 128
#define BK 32

__device__ __forceinline__ v8f wmma_bf16(const BF16x16& a, const BF16x16& b, v8f c) {
  return __builtin_amdgcn_wmma_f32_16x16x32_bf16(false, a.v, false, b.v, (short)0, c,
                                                 false, false);
}

// Async global -> LDS copy (16B per lane), tracked by ASYNCcnt.
// Generic pointer low 32 bits == LDS byte offset (aperture rule: addr[31:0]).
__device__ __forceinline__ void async_copy_b128(const __bf16* gsrc, __bf16* ldst) {
  unsigned loff = (unsigned)(unsigned long long)(uintptr_t)ldst;
  unsigned long long gaddr = (unsigned long long)(uintptr_t)gsrc;
  asm volatile("global_load_async_to_lds_b128 %0, %1, off"
               :: "v"(loff), "v"(gaddr) : "memory");
}
__device__ __forceinline__ void wait_async_all() {
  asm volatile("s_wait_asynccnt 0" ::: "memory");
}

// ---------------------------------------------------------------------------
// fp32 [K][N] -> bf16 transposed [N][K]  (LDS-tiled, done once per launch)
// ---------------------------------------------------------------------------
__global__ __launch_bounds__(256)
void transpose_to_bf16(const float* __restrict__ in, __bf16* __restrict__ out,
                       int K, int N) {
  __shared__ float tile[32][33];
  const int kb = blockIdx.x * 32, nb = blockIdx.y * 32;
  const int tx = threadIdx.x & 31, ty = threadIdx.x >> 5;  // 32 x 8
#pragma unroll
  for (int i = 0; i < 32; i += 8)
    tile[ty + i][tx] = in[(size_t)(kb + ty + i) * N + nb + tx];
  __syncthreads();
#pragma unroll
  for (int i = 0; i < 32; i += 8)
    out[(size_t)(nb + ty + i) * K + kb + tx] = (__bf16)tile[tx][ty + i];
}

// ---------------------------------------------------------------------------
// LayerNorm over C=768, one block (256 threads) per token, bf16 output
// ---------------------------------------------------------------------------
__global__ __launch_bounds__(256)
void layernorm_to_bf16(const float* __restrict__ x, const float* __restrict__ g,
                       const float* __restrict__ bet, __bf16* __restrict__ out) {
  __shared__ float rs[256];
  __shared__ float rq[256];
  const int t = blockIdx.x, tid = threadIdx.x;
  const float* row = x + (size_t)t * DIM;
  float v[3];
  float s = 0.f, q = 0.f;
#pragma unroll
  for (int i = 0; i < 3; ++i) {
    v[i] = row[tid + i * 256];
    s += v[i];
    q += v[i] * v[i];
  }
  rs[tid] = s; rq[tid] = q;
  __syncthreads();
  for (int st = 128; st > 0; st >>= 1) {
    if (tid < st) { rs[tid] += rs[tid + st]; rq[tid] += rq[tid + st]; }
    __syncthreads();
  }
  const float mean = rs[0] * (1.0f / DIM);
  const float var  = rq[0] * (1.0f / DIM) - mean * mean;
  const float rstd = rsqrtf(var + 1e-5f);
  __bf16* orow = out + (size_t)t * DIM;
#pragma unroll
  for (int i = 0; i < 3; ++i) {
    int c = tid + i * 256;
    orow[c] = (__bf16)((v[i] - mean) * rstd * g[c] + bet[c]);
  }
}

// ---------------------------------------------------------------------------
// Tiled WMMA GEMM: C[T,N] = A_bf16[T,K] @ Wt_bf16[N,K]^T + bias
//   EPI=0: store bf16; EPI=1: exact GELU -> bf16; EPI=2: +resid -> fp32
// Block = 256 threads (8 wave32). BM=BN=128, BK=32.
// Double-buffered LDS filled by async global->LDS b128 copies.
// Each wave computes a 32x64 region: 8 WMMAs / k-step, 2 A-frags x 4 B-frags.
// ---------------------------------------------------------------------------
template <int EPI>
__global__ __launch_bounds__(256)
void gemm_wmma(const __bf16* __restrict__ A, const __bf16* __restrict__ Wt,
               const float* __restrict__ bias, const float* __restrict__ resid,
               float* __restrict__ outF, __bf16* __restrict__ outB,
               int T, int K, int N) {
  __shared__ alignas(16) __bf16 As[2][BM * BK];   // [row][k]
  __shared__ alignas(16) __bf16 Bs[2][BN * BK];   // [col][k]  (Wt rows as-is)
  const int tid  = threadIdx.x;
  const int lane = tid & 31, wave = tid >> 5;
  const int hi   = lane >> 4, ln = lane & 15;
  const int m0 = blockIdx.y * BM, n0 = blockIdx.x * BN;
  const int wM = wave & 3, wN = wave >> 2;    // 4 x 2 wave grid; 32 x 64 / wave

  v8f acc[2][4];
#pragma unroll
  for (int mt = 0; mt < 2; ++mt)
#pragma unroll
    for (int nt = 0; nt < 4; ++nt) acc[mt][nt] = (v8f){};

  // Loader: 8-bf16 chunks; 4 chunks per 32-wide row; rows r0 and r0+64.
  const int r0 = tid >> 2, c0 = (tid & 3) * 8;

  auto prefetch = [&](int k0, int buf) {
    async_copy_b128(A  + (size_t)(m0 + r0)      * K + k0 + c0, &As[buf][r0 * BK + c0]);
    async_copy_b128(A  + (size_t)(m0 + r0 + 64) * K + k0 + c0, &As[buf][(r0 + 64) * BK + c0]);
    async_copy_b128(Wt + (size_t)(n0 + r0)      * K + k0 + c0, &Bs[buf][r0 * BK + c0]);
    async_copy_b128(Wt + (size_t)(n0 + r0 + 64) * K + k0 + c0, &Bs[buf][(r0 + 64) * BK + c0]);
  };

  prefetch(0, 0);
  wait_async_all();
  __syncthreads();

  int cur = 0;
  for (int k0 = 0; k0 < K; k0 += BK) {
    if (k0 + BK < K) prefetch(k0 + BK, cur ^ 1);

    const unsigned* Au = (const unsigned*)As[cur];
    const unsigned* Bu = (const unsigned*)Bs[cur];
    BF16x16 af[2], bf[4];
#pragma unroll
    for (int mt = 0; mt < 2; ++mt) {              // A frag: K = hi*8+{0..7}, +16
      const int r = wM * 32 + mt * 16 + ln;
#pragma unroll
      for (int i = 0; i < 4; ++i) {
        af[mt].u[i]     = Au[r * 16 + hi * 4 + i];
        af[mt].u[4 + i] = Au[r * 16 + hi * 4 + 8 + i];
      }
    }
#pragma unroll
    for (int nt = 0; nt < 4; ++nt) {              // B frag: K = hi*8+{0..7}, +16
      const int c = wN * 64 + nt * 16 + ln;
#pragma unroll
      for (int i = 0; i < 8; ++i) bf[nt].u[i] = Bu[c * 16 + hi * 4 + i + (i & 4)];
    }
#pragma unroll
    for (int mt = 0; mt < 2; ++mt)
#pragma unroll
      for (int nt = 0; nt < 4; ++nt)
        acc[mt][nt] = wmma_bf16(af[mt], bf[nt], acc[mt][nt]);

    wait_async_all();
    __syncthreads();
    cur ^= 1;
  }

#pragma unroll
  for (int mt = 0; mt < 2; ++mt)
#pragma unroll
    for (int nt = 0; nt < 4; ++nt) {
      const int col = n0 + wN * 64 + nt * 16 + ln;
      const float bv = bias[col];
#pragma unroll
      for (int j = 0; j < 8; ++j) {
        const int row = m0 + wM * 32 + mt * 16 + j + hi * 8;
        const size_t idx = (size_t)row * N + col;
        float v = acc[mt][nt][j] + bv;
        if (EPI == 0) {
          outB[idx] = (__bf16)v;
        } else if (EPI == 1) {
          v = 0.5f * v * (1.0f + erff(v * 0.70710678118654752f));  // exact GELU
          outB[idx] = (__bf16)v;
        } else {
          outF[idx] = resid[idx] + v;
        }
      }
    }
}

// ---------------------------------------------------------------------------
// Flash attention: grid (NCTX/128, NH, B), 256 threads = 8 waves.
// Each wave owns a 16-row Q tile; block shares 32-key K/V tiles in LDS.
// ---------------------------------------------------------------------------
__global__ __launch_bounds__(256)
void attn_wmma(const __bf16* __restrict__ qkv, __bf16* __restrict__ ctx) {
  __shared__ alignas(16) __bf16 Ks[32 * 64];       // [key][d]   == B^t for S
  __shared__ alignas(16) __bf16 Vt[64 * 32];       // [d][key]   == B^t for P@V
  __shared__ alignas(16) __bf16 Ps[8 * 16 * 32];   // per-wave P tile
  const int tid = threadIdx.x, lane = tid & 31, wave = tid >> 5;
  const int hi = lane >> 4, ln = lane & 15;
  const int b = blockIdx.z, h = blockIdx.y;
  const int q0 = blockIdx.x * 128 + wave * 16;
  const size_t rowbase = ((size_t)(b * NCTX + q0 + ln)) * (3 * DIM) + h * HD;

  // Q fragments (two K-chunks of 32 over d=64), straight from global in A-layout
  BF16x16 qf[2];
  {
    const unsigned* qp = (const unsigned*)(qkv + rowbase);
#pragma unroll
    for (int c = 0; c < 2; ++c)
#pragma unroll
      for (int i = 0; i < 4; ++i) {
        qf[c].u[i]     = qp[c * 16 + hi * 4 + i];
        qf[c].u[4 + i] = qp[c * 16 + hi * 4 + 8 + i];
      }
  }

  float mrow[8], lrow[8];
  v8f acc[4];
#pragma unroll
  for (int j = 0; j < 8; ++j) { mrow[j] = -1e30f; lrow[j] = 0.f; }
#pragma unroll
  for (int d = 0; d < 4; ++d) acc[d] = (v8f){};

  const int kr = tid >> 3, dc = (tid & 7) * 8;
  const float scale = 0.125f;  // 1/sqrt(64)

  for (int kt = 0; kt < NCTX; kt += 32) {
    __syncthreads();
    {
      const size_t srow = ((size_t)(b * NCTX + kt + kr)) * (3 * DIM) + h * HD + dc;
      *(uint4*)(Ks + kr * 64 + dc) = *(const uint4*)(qkv + srow + DIM);       // K
      uint4 vv = *(const uint4*)(qkv + srow + 2 * DIM);                       // V
      const __bf16* vp = (const __bf16*)&vv;
#pragma unroll
      for (int j = 0; j < 8; ++j) Vt[(dc + j) * 32 + kr] = vp[j];
    }
    __syncthreads();

    // ---- S = Q @ K^T for two 16-key subtiles ----
    const unsigned* Ku = (const unsigned*)Ks;
    v8f s0 = (v8f){}, s1 = (v8f){};
#pragma unroll
    for (int c = 0; c < 2; ++c) {
      BF16x16 kb0, kb1;
#pragma unroll
      for (int i = 0; i < 8; ++i) {
        kb0.u[i] = Ku[(ln)      * 32 + c * 16 + hi * 4 + i + (i & 4)];
        kb1.u[i] = Ku[(16 + ln) * 32 + c * 16 + hi * 4 + i + (i & 4)];
      }
      s0 = wmma_bf16(qf[c], kb0, s0);
      s1 = wmma_bf16(qf[c], kb1, s1);
    }

    // ---- online softmax (fp32, 16-lane row reductions on wave32) ----
    float p0[8], p1[8];
#pragma unroll
    for (int j = 0; j < 8; ++j) {
      float a = s0[j] * scale, bb = s1[j] * scale;
      float r = fmaxf(a, bb);
      r = fmaxf(r, __shfl_xor(r, 1, 16));
      r = fmaxf(r, __shfl_xor(r, 2, 16));
      r = fmaxf(r, __shfl_xor(r, 4, 16));
      r = fmaxf(r, __shfl_xor(r, 8, 16));
      const float mnew = fmaxf(mrow[j], r);
      const float al = __expf(mrow[j] - mnew);
      const float e0 = __expf(a - mnew), e1 = __expf(bb - mnew);
      float rsum = e0 + e1;
      rsum += __shfl_xor(rsum, 1, 16);
      rsum += __shfl_xor(rsum, 2, 16);
      rsum += __shfl_xor(rsum, 4, 16);
      rsum += __shfl_xor(rsum, 8, 16);
      lrow[j] = lrow[j] * al + rsum;
      mrow[j] = mnew;
      p0[j] = e0; p1[j] = e1;
#pragma unroll
      for (int d = 0; d < 4; ++d) acc[d][j] *= al;
    }

    // ---- P (D-layout) -> per-wave LDS -> A-layout fragment ----
    __bf16* Pw = Ps + wave * (16 * 32);
#pragma unroll
    for (int j = 0; j < 8; ++j) {
      const int r = j + hi * 8;
      Pw[r * 32 + ln]      = (__bf16)p0[j];
      Pw[r * 32 + 16 + ln] = (__bf16)p1[j];
    }
    asm volatile("s_wait_dscnt 0" ::: "memory");  // wave-local DS RAW fence
    BF16x16 pf;
    const unsigned* Pu = (const unsigned*)Pw;
#pragma unroll
    for (int i = 0; i < 4; ++i) {
      pf.u[i]     = Pu[ln * 16 + hi * 4 + i];
      pf.u[4 + i] = Pu[ln * 16 + hi * 4 + 8 + i];
    }

    // ---- ctx += P @ V over four 16-wide d subtiles ----
    const unsigned* Vu = (const unsigned*)Vt;
#pragma unroll
    for (int d = 0; d < 4; ++d) {
      BF16x16 vb;
#pragma unroll
      for (int i = 0; i < 8; ++i) vb.u[i] = Vu[(d * 16 + ln) * 16 + hi * 4 + i + (i & 4)];
      acc[d] = wmma_bf16(pf, vb, acc[d]);
    }
  }

  // ---- normalize and store bf16 ctx ----
#pragma unroll
  for (int j = 0; j < 8; ++j) {
    const float inv = 1.0f / lrow[j];
    const int row = q0 + j + hi * 8;
    __bf16* op = ctx + ((size_t)(b * NCTX + row)) * DIM + h * HD;
#pragma unroll
    for (int d = 0; d < 4; ++d) op[d * 16 + ln] = (__bf16)(acc[d][j] * inv);
  }
}

// ---------------------------------------------------------------------------
extern "C" void kernel_launch(void* const* d_in, const int* in_sizes, int n_in,
                              void* d_out, int out_size, void* d_ws, size_t ws_size,
                              hipStream_t stream) {
  const float* x      = (const float*)d_in[0];
  const float* ln1_g  = (const float*)d_in[1];
  const float* ln1_b  = (const float*)d_in[2];
  const float* qkv_w  = (const float*)d_in[3];
  const float* qkv_b  = (const float*)d_in[4];
  const float* proj_w = (const float*)d_in[5];
  const float* proj_b = (const float*)d_in[6];
  const float* ln2_g  = (const float*)d_in[7];
  const float* ln2_b  = (const float*)d_in[8];
  const float* w1     = (const float*)d_in[9];
  const float* b1     = (const float*)d_in[10];
  const float* w2     = (const float*)d_in[11];
  const float* b2     = (const float*)d_in[12];
  float* out = (float*)d_out;

  const int T  = in_sizes[0] / DIM;  // B*N tokens (4096)
  const int Bn = T / NCTX;           // batch (2)

  // workspace carve-up (256B aligned)
  char* wp = (char*)d_ws;
  auto carve = [&](size_t bytes) {
    char* p = wp;
    wp += (bytes + 255) & ~(size_t)255;
    return p;
  };
  __bf16* qkv_wt  = (__bf16*)carve((size_t)DIM * 3 * DIM * 2);   // [3*DIM][DIM]
  __bf16* proj_wt = (__bf16*)carve((size_t)DIM * DIM * 2);       // [DIM][DIM]
  __bf16* w1t     = (__bf16*)carve((size_t)DIM * HIDDEN * 2);    // [HIDDEN][DIM]
  __bf16* w2t     = (__bf16*)carve((size_t)HIDDEN * DIM * 2);    // [DIM][HIDDEN]
  __bf16* h1      = (__bf16*)carve((size_t)T * DIM * 2);
  __bf16* qkvb    = (__bf16*)carve((size_t)T * 3 * DIM * 2);
  __bf16* ctxb    = (__bf16*)carve((size_t)T * DIM * 2);
  float*  x1      = (float*) carve((size_t)T * DIM * 4);
  __bf16* h2      = (__bf16*)carve((size_t)T * DIM * 2);
  __bf16* hid     = (__bf16*)carve((size_t)T * HIDDEN * 2);

  // 1) weights -> bf16, transposed to [N][K]
  transpose_to_bf16<<<dim3(DIM / 32, 3 * DIM / 32), 256, 0, stream>>>(
      qkv_w, qkv_wt, DIM, 3 * DIM);
  transpose_to_bf16<<<dim3(DIM / 32, DIM / 32), 256, 0, stream>>>(
      proj_w, proj_wt, DIM, DIM);
  transpose_to_bf16<<<dim3(DIM / 32, HIDDEN / 32), 256, 0, stream>>>(
      w1, w1t, DIM, HIDDEN);
  transpose_to_bf16<<<dim3(HIDDEN / 32, DIM / 32), 256, 0, stream>>>(
      w2, w2t, HIDDEN, DIM);

  // 2) ln1
  layernorm_to_bf16<<<T, 256, 0, stream>>>(x, ln1_g, ln1_b, h1);

  // 3) qkv = h1 @ qkv_w + qkv_b  (bf16 out)
  gemm_wmma<0><<<dim3(3 * DIM / BN, T / BM), 256, 0, stream>>>(
      h1, qkv_wt, qkv_b, nullptr, nullptr, qkvb, T, DIM, 3 * DIM);

  // 4) attention -> ctx (bf16)
  attn_wmma<<<dim3(NCTX / 128, NH, Bn), 256, 0, stream>>>(qkvb, ctxb);

  // 5) x1 = x + ctx @ proj_w + proj_b  (fp32 out)
  gemm_wmma<2><<<dim3(DIM / BN, T / BM), 256, 0, stream>>>(
      ctxb, proj_wt, proj_b, x, x1, nullptr, T, DIM, DIM);

  // 6) ln2
  layernorm_to_bf16<<<T, 256, 0, stream>>>(x1, ln2_g, ln2_b, h2);

  // 7) hid = gelu(h2 @ w1 + b1)  (bf16 out)
  gemm_wmma<1><<<dim3(HIDDEN / BN, T / BM), 256, 0, stream>>>(
      h2, w1t, b1, nullptr, nullptr, hid, T, DIM, HIDDEN);

  // 8) out = x1 + hid @ w2 + b2  (fp32 out)
  gemm_wmma<2><<<dim3(DIM / BN, T / BM), 256, 0, stream>>>(
      hid, w2t, b2, x1, out, nullptr, T, HIDDEN, DIM);

  (void)n_in; (void)out_size; (void)ws_size;
}
